// Retriever_28046136443213
// MI455X (gfx1250) — matrix operand order repaired and checked by reading, not compiled, not run on hardware
//
#include <hip/hip_runtime.h>
#include <hip/hip_bf16.h>
#include <math.h>

typedef __attribute__((ext_vector_type(16))) __bf16 v16bf;
typedef __attribute__((ext_vector_type(8)))  __bf16 v8bf;
typedef __attribute__((ext_vector_type(8)))  float  v8f;

#define DIM     768
#define NQ      64
#define TILE_N  128
#define KSTEP   32
#define NKSTEP  (DIM / KSTEP)   // 24
#define TOPK    5
#define BSTRIDE 40              // padded LDS row stride (bf16 elems) for B tile
#define SSTRIDE 132             // padded LDS row stride (f32) for sims tile
#define NEG_INF (-3.0e38f)

// ---------------------------------------------------------------------------
// Kernel 1: L2-normalize queries (fp32 math) and emit bf16 Q into workspace.
// One block per query row; 768 = 256 threads x 3 elements.
// ---------------------------------------------------------------------------
__global__ __launch_bounds__(256) void qnorm_kernel(const float* __restrict__ q,
                                                    __bf16* __restrict__ qn) {
  __shared__ float red[8];
  __shared__ float s_inv;
  const int row = blockIdx.x;
  const int t   = threadIdx.x;
  const float x0 = q[row * DIM + t];
  const float x1 = q[row * DIM + t + 256];
  const float x2 = q[row * DIM + t + 512];
  float ss = x0 * x0 + x1 * x1 + x2 * x2;
#pragma unroll
  for (int m = 16; m >= 1; m >>= 1) ss += __shfl_xor(ss, m, 32);
  if ((t & 31) == 0) red[t >> 5] = ss;
  __syncthreads();
  if (t == 0) {
    float s = 0.f;
#pragma unroll
    for (int i = 0; i < 8; ++i) s += red[i];
    s_inv = 1.0f / fmaxf(sqrtf(s), 1e-12f);
  }
  __syncthreads();
  const float inv = s_inv;
  qn[row * DIM + t]       = (__bf16)(x0 * inv);
  qn[row * DIM + t + 256] = (__bf16)(x1 * inv);
  qn[row * DIM + t + 512] = (__bf16)(x2 * inv);
}

__device__ __forceinline__ void topk_insert(float v, int id, float tv[TOPK], int ti[TOPK]) {
  if (v <= tv[TOPK - 1]) return;
  tv[TOPK - 1] = v;
  ti[TOPK - 1] = id;
#pragma unroll
  for (int j = TOPK - 1; j > 0; --j) {
    if (tv[j] > tv[j - 1]) {
      float fv = tv[j]; tv[j] = tv[j - 1]; tv[j - 1] = fv;
      int   fi = ti[j]; ti[j] = ti[j - 1]; ti[j - 1] = fi;
    }
  }
}

// ---------------------------------------------------------------------------
// Kernel 2: fused bf16-WMMA GEMM (Qn x bank^T), bank-norm rsqrt fused into the
// tile load, per-block running top-5 per query. Grid-stride over 128-col tiles.
// ---------------------------------------------------------------------------
__global__ __launch_bounds__(256) void sim_topk_kernel(const float* __restrict__ bank,
                                                       const __bf16* __restrict__ qn,
                                                       float* __restrict__ cand_vals,
                                                       int* __restrict__ cand_ids,
                                                       int N) {
  __shared__ __bf16 lds_b[TILE_N * BSTRIDE];     // bank tile, bf16, n-major
  __shared__ float  lds_inv[TILE_N];             // 1/||bank_row||
  __shared__ float  lds_sims[NQ * SSTRIDE];      // 64 x 128 sims (padded)

  const int t  = threadIdx.x;
  const int l  = t & 31;          // lane
  const int w  = t >> 5;          // wave 0..7
  const int mt = w & 3;           // M tile 0..3 (rows mt*16..+15)
  const int ng = w >> 2;          // N group 0..1 (cols ng*64..+63)

  // A-fragment addressing (16x32 bf16): lane<16 holds K {0..7,16..23}, lane>=16 {8..15,24..31}
  const int   arow  = mt * 16 + (l & 15);
  const int   koffA = (l >> 4) * 8;
  const __bf16* qrow = qn + arow * DIM;
  // B-fragment addressing (32x16 bf16): lane<16 holds K {0..7,16..23}, lane>=16 {8..15,24..31}
  const int koffB = (l >> 4) * 8;

  // Per-thread bank-tile load role: row n_loc = t/2, k-half = (t&1)*16
  const int n_loc = t >> 1;
  const int khalf = (t & 1) * 16;

  float tv[TOPK];
  int   ti[TOPK];
#pragma unroll
  for (int j = 0; j < TOPK; ++j) { tv[j] = NEG_INF; ti[j] = -1; }

  for (int tile = blockIdx.x; tile * TILE_N < N; tile += gridDim.x) {
    const int  n_glob = tile * TILE_N + n_loc;
    const bool n_ok   = n_glob < N;
    const float* brow = bank + (size_t)n_glob * DIM;

    // Speculative prefetch of this thread's row for the NEXT tile.
    {
      long nn = (long)(tile + gridDim.x) * TILE_N + n_loc;
      if (nn < N) __builtin_prefetch(bank + (size_t)nn * DIM + khalf, 0, 1);
    }

    v8f acc0 = {}, acc1 = {}, acc2 = {}, acc3 = {};
    float ss = 0.f;  // sum of squares for row n_loc (this thread's k-half)

    for (int kk = 0; kk < NKSTEP; ++kk) {
      const int k0 = kk * KSTEP;
      // --- load 16 fp32, accumulate norms, convert to bf16, stage in LDS ---
      float f[16];
      if (n_ok) {
#pragma unroll
        for (int i = 0; i < 4; ++i) {
          float4 v = *(const float4*)(brow + k0 + khalf + 4 * i);
          f[4 * i + 0] = v.x; f[4 * i + 1] = v.y; f[4 * i + 2] = v.z; f[4 * i + 3] = v.w;
        }
      } else {
#pragma unroll
        for (int i = 0; i < 16; ++i) f[i] = 0.f;
      }
      v8bf p0, p1;
#pragma unroll
      for (int i = 0; i < 8; ++i) {
        ss += f[i] * f[i];
        ss += f[i + 8] * f[i + 8];
        p0[i] = (__bf16)f[i];
        p1[i] = (__bf16)f[i + 8];
      }
      __bf16* wp = &lds_b[n_loc * BSTRIDE + khalf];
      *(v8bf*)(wp)     = p0;
      *(v8bf*)(wp + 8) = p1;
      __syncthreads();

      // --- fragments + 4 WMMAs (uniform control flow: EXEC all ones) ---
      v8bf alo = *(const v8bf*)(qrow + k0 + koffA);
      v8bf ahi = *(const v8bf*)(qrow + k0 + koffA + 16);
      v16bf a = __builtin_shufflevector(alo, ahi, 0,1,2,3,4,5,6,7,8,9,10,11,12,13,14,15);
#pragma unroll
      for (int j = 0; j < 4; ++j) {
        const __bf16* bp = &lds_b[(ng * 64 + j * 16 + (l & 15)) * BSTRIDE + koffB];
        v8bf blo = *(const v8bf*)(bp);
        v8bf bhi = *(const v8bf*)(bp + 16);
        v16bf b = __builtin_shufflevector(blo, bhi, 0,1,2,3,4,5,6,7,8,9,10,11,12,13,14,15);
        if (j == 0) acc0 = __builtin_amdgcn_wmma_f32_16x16x32_bf16(false, a, false, b, (short)0, acc0, false, false);
        if (j == 1) acc1 = __builtin_amdgcn_wmma_f32_16x16x32_bf16(false, a, false, b, (short)0, acc1, false, false);
        if (j == 2) acc2 = __builtin_amdgcn_wmma_f32_16x16x32_bf16(false, a, false, b, (short)0, acc2, false, false);
        if (j == 3) acc3 = __builtin_amdgcn_wmma_f32_16x16x32_bf16(false, a, false, b, (short)0, acc3, false, false);
      }
      __syncthreads();
    }

    // --- finish bank-row norms: combine the two k-half partials per row ---
    float ss_tot = ss + __shfl_xor(ss, 1, 32);
    if ((t & 1) == 0) lds_inv[n_loc] = 1.0f / fmaxf(sqrtf(ss_tot), 1e-12f);
    __syncthreads();

    // --- scale + scatter sims to LDS (D layout: M=(l/16)*8+v, N=l&15) ---
#pragma unroll
    for (int j = 0; j < 4; ++j) {
#pragma unroll
      for (int v = 0; v < 8; ++v) {
        const int  row  = mt * 16 + (l >> 4) * 8 + v;
        const int  coll = ng * 64 + j * 16 + (l & 15);
        const long gcol = (long)tile * TILE_N + coll;
        float sv = (j == 0 ? acc0[v] : j == 1 ? acc1[v] : j == 2 ? acc2[v] : acc3[v]);
        sv *= lds_inv[coll];
        if (gcol >= N) sv = NEG_INF;
        lds_sims[row * SSTRIDE + coll] = sv;
      }
    }
    __syncthreads();

    // --- threads 0..63: fold this tile into the running per-query top-5 ---
    if (t < NQ) {
      const int gbase = tile * TILE_N;
      for (int c = 0; c < TILE_N; ++c)
        topk_insert(lds_sims[t * SSTRIDE + c], gbase + c, tv, ti);
    }
  }

  if (t < NQ) {
    const int base = (blockIdx.x * NQ + t) * TOPK;
#pragma unroll
    for (int j = 0; j < TOPK; ++j) { cand_vals[base + j] = tv[j]; cand_ids[base + j] = ti[j]; }
  }
}

// ---------------------------------------------------------------------------
// Kernel 3: merge per-block candidates into final top-5; one block per query.
// ---------------------------------------------------------------------------
__global__ __launch_bounds__(128) void merge_kernel(const float* __restrict__ cand_vals,
                                                    const int* __restrict__ cand_ids,
                                                    float* __restrict__ out,
                                                    int nblk) {
  __shared__ float mv[128 * TOPK];
  __shared__ int   mi[128 * TOPK];
  const int q = blockIdx.x;
  const int t = threadIdx.x;
  float tv[TOPK]; int ti[TOPK];
#pragma unroll
  for (int j = 0; j < TOPK; ++j) { tv[j] = NEG_INF; ti[j] = -1; }
  const int nc = nblk * TOPK;
  for (int c = t; c < nc; c += 128) {
    const int b = c / TOPK, j = c % TOPK;
    const int idx = (b * NQ + q) * TOPK + j;
    topk_insert(cand_vals[idx], cand_ids[idx], tv, ti);
  }
#pragma unroll
  for (int j = 0; j < TOPK; ++j) { mv[t * TOPK + j] = tv[j]; mi[t * TOPK + j] = ti[j]; }
  __syncthreads();
  if (t == 0) {
    float fv[TOPK]; int fi[TOPK];
#pragma unroll
    for (int j = 0; j < TOPK; ++j) { fv[j] = NEG_INF; fi[j] = -1; }
    for (int c = 0; c < 128 * TOPK; ++c) topk_insert(mv[c], mi[c], fv, fi);
#pragma unroll
    for (int j = 0; j < TOPK; ++j) {
      out[q * TOPK + j]              = fv[j];
      out[NQ * TOPK + q * TOPK + j] = (float)fi[j];  // ids exact in f32 (< 2^24)
    }
  }
}

// ---------------------------------------------------------------------------
extern "C" void kernel_launch(void* const* d_in, const int* in_sizes, int n_in,
                              void* d_out, int out_size, void* d_ws, size_t ws_size,
                              hipStream_t stream) {
  const float* query = (const float*)d_in[0];
  const float* bank  = (const float*)d_in[1];
  const int N = in_sizes[1] / DIM;  // 500000

  // Workspace layout: [bf16 Qn: 64*768*2 = 98304 B][cand_vals][cand_ids]
  char* ws = (char*)d_ws;
  __bf16* qn = (__bf16*)ws;
  const size_t qn_bytes = (size_t)NQ * DIM * sizeof(__bf16);
  size_t avail = ws_size > qn_bytes ? ws_size - qn_bytes : 0;
  int nblk = (int)(avail / ((size_t)NQ * TOPK * (sizeof(float) + sizeof(int))));
  if (nblk > 512) nblk = 512;
  if (nblk < 1)  nblk = 1;
  float* cand_vals = (float*)(ws + qn_bytes);
  int*   cand_ids  = (int*)(cand_vals + (size_t)nblk * NQ * TOPK);

  qnorm_kernel<<<NQ, 256, 0, stream>>>(query, qn);
  sim_topk_kernel<<<nblk, 256, 0, stream>>>(bank, qn, cand_vals, cand_ids, N);
  merge_kernel<<<NQ, 128, 0, stream>>>(cand_vals, cand_ids, (float*)d_out, nblk);
}
